// ResidualLSTM_24601572671977
// MI455X (gfx1250) — matrix operand order repaired and checked by reading, not compiled, not run on hardware
//
#include <hip/hip_runtime.h>
#include <hip/hip_bf16.h>
#include <math.h>

// Problem constants (from reference)
#define RB   512
#define RT   1024
#define RIN  64
#define ROUT 32
#define RH   128
#define RL   4

typedef __attribute__((ext_vector_type(16))) _Float16 v16h;
typedef __attribute__((ext_vector_type(8)))  _Float16 v8h;
typedef __attribute__((ext_vector_type(8)))  float    v8f;

union V16U { v16h v; v8h h[2]; };

__device__ __forceinline__ v8f wmma16(v16h a, v16h b, v8f c) {
  // v_wmma_f32_16x16x32_f16: D = A(16x32 f16) x B(32x16 f16) + C(16x16 f32)
  return __builtin_amdgcn_wmma_f32_16x16x32_f16(false, a, false, b, (short)0, c,
                                                false, false);
}

// A-fragment (16x32 f16, row-major source with row stride `ld` halves).
// ISA layout: lanes 0-15 hold M=lane, K={0..7,16..23}; lanes 16-31 hold
// M=lane-16, K={8..15,24..31}.
__device__ __forceinline__ v16h load_a16(const _Float16* s, int ld) {
  const int lane = threadIdx.x & 31;
  const _Float16* row = s + (size_t)(lane & 15) * ld + ((lane < 16) ? 0 : 8);
  V16U u;
  u.h[0] = *(const v8h*)(row);
  u.h[1] = *(const v8h*)(row + 16);
  return u.v;
}

// B-fragment from a pre-packed 512-half tile blob: lane holds N=lane%16,
// 16 contiguous halves = K 8h+0..7, 16+8h+0..7 (h = lane/16)  (two global_load_b128 per lane).
__device__ __forceinline__ v16h load_b16(const _Float16* tile) {
  const int lane = threadIdx.x & 31;
  return *(const v16h*)(tile + lane * 16);
}

__device__ __forceinline__ float sigmf(float x) { return 1.0f / (1.0f + expf(-x)); }

// Generic shared pointer -> 32-bit LDS byte offset (flat LDS aperture uses
// addr[31:0] directly).
__device__ __forceinline__ uint32_t lds_off(const void* p) {
  return (uint32_t)(uintptr_t)p;
}

// CDNA5 async DMA: copy 16 bytes global -> LDS per lane, tracked by ASYNCcnt.
__device__ __forceinline__ void async_b128(uint32_t lds, const void* gaddr) {
  asm volatile("global_load_async_to_lds_b128 %0, %1, off"
               :: "v"(lds), "v"(gaddr) : "memory");
}
__device__ __forceinline__ void wait_async_le1() {
  asm volatile("s_wait_asynccnt 0x1" ::: "memory");
}
__device__ __forceinline__ void wait_async_0() {
  asm volatile("s_wait_asynccnt 0x0" ::: "memory");
}

// -------------------------------------------------------------------------
// Pack W [N,K] f32 (gates = inp @ W^T, so B[k,n] = W[n,k]) into B-fragment
// tile blobs: tile (nt,kt) at dst[(nt*KT+kt)*512], element [lane*16+j] =
// W[nt*16+(lane&15), kt*32+(j&7)+8*(lane>>4)+16*(j>>3)], converted to f16.
// -------------------------------------------------------------------------
__global__ void pack_b_kernel(const float* __restrict__ W,
                              _Float16* __restrict__ dst, int N, int K) {
  const int tile = blockIdx.x;
  const int KT = K >> 5;
  const int nt = tile / KT, kt = tile % KT;
  const int n0 = nt * 16, k0 = kt * 32;
  for (int e = threadIdx.x; e < 512; e += 256) {
    const int lane = e >> 4, j = e & 15;
    dst[(size_t)tile * 512 + e] =
        (_Float16)W[(size_t)(n0 + (lane & 15)) * K + (k0 + (j & 7) + 8 * (lane >> 4) + 16 * (j >> 3))];
  }
}

__global__ void bias_sum_kernel(const float* __restrict__ a,
                                const float* __restrict__ b,
                                float* __restrict__ s, int n) {
  const int i = blockIdx.x * 256 + threadIdx.x;
  if (i < n) s[i] = a[i] + b[i];
}

// -------------------------------------------------------------------------
// Input projection: xp[t,b,h] = sum_i x[b,t,i]*in_W[h,i] + in_b[h], f16 out.
// One block = 16 rows (one t, 16 consecutive b) x 128 cols; 8 waves, wave w
// owns N-tile w; K = 64 -> 2 k-steps.
// -------------------------------------------------------------------------
__global__ void __launch_bounds__(256) inproj_kernel(
    const float* __restrict__ x, const _Float16* __restrict__ inWp,
    const float* __restrict__ in_b, _Float16* __restrict__ xp) {
  __shared__ __align__(16) _Float16 a16[16][RIN];
  const int g = blockIdx.x;
  const int BT16 = RB / 16;               // 32
  const int t = g / BT16;
  const int b0 = (g % BT16) * 16;

  for (int e = threadIdx.x; e < 16 * RIN; e += 256) {
    const int r = e >> 6, c = e & 63;
    a16[r][c] = (_Float16)x[((size_t)(b0 + r) * RT + t) * RIN + c];
  }
  __syncthreads();

  const int wave = threadIdx.x >> 5, lane = threadIdx.x & 31;
  v8f acc = {};
#pragma unroll
  for (int kt = 0; kt < 2; ++kt) {
    v16h a = load_a16(&a16[0][0] + kt * 32, RIN);
    v16h b = load_b16(inWp + (size_t)(wave * 2 + kt) * 512);
    acc = wmma16(a, b, acc);
  }
  const int nn = lane & 15;
  const int n = wave * 16 + nn;
  const int mb = (lane < 16) ? 0 : 8;
  const float bias = in_b[n];
#pragma unroll
  for (int v = 0; v < 8; ++v) {
    xp[((size_t)t * RB + b0 + mb + v) * RH + n] = (_Float16)(acc[v] + bias);
  }
}

// -------------------------------------------------------------------------
// Persistent LSTM scan. 32 blocks x 256 threads; block owns 16 batch rows
// for the entire T loop.
//  - xp tile double-buffered in LDS via global_load_async_to_lds_b128,
//    prefetched one step ahead (ASYNCcnt-gated) -> off the critical path.
//  - Per layer: gates[16,512] = inp@Wih^T + h@Whh^T, 32 WMMAs/wave.
//  - Elementwise: row = 16 consecutive lanes -> LayerNorm reduction is an
//    intra-wave shfl_xor tree; c state lives in registers.
// -------------------------------------------------------------------------
__global__ void __launch_bounds__(256, 1) lstm_scan_kernel(
    const _Float16* __restrict__ xp, const _Float16* __restrict__ wihp,
    const _Float16* __restrict__ whhp, const float* __restrict__ bsum,
    const float* __restrict__ lng, const float* __restrict__ lnb,
    _Float16* __restrict__ last16) {
  __shared__ __align__(16) _Float16 h16[RL][16][RH];    // 16 KB f16 h state
  __shared__ __align__(16) _Float16 xbuf[2][16][RH];    // 8 KB xp double buffer
  __shared__ __align__(16) float    inpf[16][RH];       // 8 KB layer input f32
  __shared__ __align__(16) float    gates[16][4 * RH];  // 32 KB

  const int tid = threadIdx.x;
  const int wave = tid >> 5, lane = tid & 31;
  const int b0 = blockIdx.x * 16;
  const int r_ew = tid >> 4, cg = tid & 15;  // row = 16 consecutive lanes

  for (int e = tid; e < RL * 16 * RH; e += 256) ((_Float16*)h16)[e] = (_Float16)0.0f;
  float creg[RL][8];  // cell state, thread-private (row r_ew, cols cg*8..+7)
#pragma unroll
  for (int l = 0; l < RL; ++l)
#pragma unroll
    for (int u = 0; u < 8; ++u) creg[l][u] = 0.0f;

  const int mb = (lane < 16) ? 0 : 8;
  const int nn = lane & 15;

  // Async prefetch of tile 0 (each lane moves 16 bytes).
  const uint32_t xls0 = lds_off(&xbuf[0][0][0]) + (uint32_t)tid * 16;
  const uint32_t xls1 = lds_off(&xbuf[1][0][0]) + (uint32_t)tid * 16;
  async_b128(xls0, xp + (size_t)b0 * RH + tid * 8);

  for (int t = 0; t < RT; ++t) {
    if (t + 1 < RT) {
      async_b128((((t + 1) & 1) ? xls1 : xls0),
                 xp + ((size_t)(t + 1) * RB + b0) * RH + tid * 8);
      wait_async_le1();  // tile t complete (in-order), t+1 still in flight
    } else {
      wait_async_0();
    }
    __syncthreads();

    for (int layer = 0; layer < RL; ++layer) {
      // ---- GEMM: gates = inp @ Wih^T + h_old @ Whh^T --------------------
      const _Float16* ainp =
          (layer == 0) ? &xbuf[t & 1][0][0] : &h16[layer - 1][0][0];
      const _Float16* wih = wihp + (size_t)layer * 32 * 4 * 512;
      const _Float16* whh = whhp + (size_t)layer * 32 * 4 * 512;
      v8f acc[4];
#pragma unroll
      for (int q = 0; q < 4; ++q) acc[q] = (v8f){};
#pragma unroll
      for (int kt = 0; kt < 4; ++kt) {
        v16h ai = load_a16(ainp + kt * 32, RH);
        v16h ah = load_a16(&h16[layer][0][0] + kt * 32, RH);
#pragma unroll
        for (int q = 0; q < 4; ++q) {
          const int nt = wave * 4 + q;
          v16h bi = load_b16(wih + (size_t)(nt * 4 + kt) * 512);
          v16h bh = load_b16(whh + (size_t)(nt * 4 + kt) * 512);
          acc[q] = wmma16(ai, bi, acc[q]);
          acc[q] = wmma16(ah, bh, acc[q]);
        }
      }
#pragma unroll
      for (int q = 0; q < 4; ++q) {
        const int n0 = (wave * 4 + q) * 16;
#pragma unroll
        for (int v = 0; v < 8; ++v) gates[mb + v][n0 + nn] = acc[q][v];
      }
      __syncthreads();

      // ---- elementwise: LSTM cell + residual + LayerNorm ----------------
      const float* bs = bsum + layer * 4 * RH;
      float s = 0.0f, s2 = 0.0f;
      float zv[8];
#pragma unroll
      for (int u = 0; u < 8; ++u) {
        const int j = cg * 8 + u;  // 8 contiguous columns per thread
        const float ig = gates[r_ew][j] + bs[j];
        const float fg = gates[r_ew][RH + j] + bs[RH + j];
        const float gg = gates[r_ew][2 * RH + j] + bs[2 * RH + j];
        const float og = gates[r_ew][3 * RH + j] + bs[3 * RH + j];
        const float ci = sigmf(fg) * creg[layer][u] + sigmf(ig) * tanhf(gg);
        creg[layer][u] = ci;
        const float hp = sigmf(og) * tanhf(ci);
        const float z = (layer > 0) ? hp + inpf[r_ew][j] : hp;
        zv[u] = z;
        s += z;
        s2 += z * z;
      }
      // Row reduction: the 16 workers of a row are lanes (base..base+15) of
      // one wave -> xor-shuffle tree stays inside the group.
#pragma unroll
      for (int m = 1; m <= 8; m <<= 1) {
        s += __shfl_xor(s, m, 32);
        s2 += __shfl_xor(s2, m, 32);
      }
      const float mu = s * (1.0f / 128.0f);
      const float var = s2 * (1.0f / 128.0f) - mu * mu;
      const float rs = rsqrtf(var + 1e-5f);
#pragma unroll
      for (int u = 0; u < 8; ++u) {
        const int j = cg * 8 + u;
        const float hi =
            (zv[u] - mu) * rs * lng[layer * RH + j] + lnb[layer * RH + j];
        inpf[r_ew][j] = hi;
        h16[layer][r_ew][j] = (_Float16)hi;
      }
      __syncthreads();
    }
  }
  // h16[L-1] holds h of the last layer at t = T-1 (f16).
  ((uint4*)(last16 + (size_t)b0 * RH))[tid] =
      ((const uint4*)(&h16[RL - 1][0][0]))[tid];
}

// -------------------------------------------------------------------------
// Head: out = relu(last @ W1^T + b1) @ W2^T + b2.  32 blocks x 256 threads.
// -------------------------------------------------------------------------
__global__ void __launch_bounds__(256) head_kernel(
    const _Float16* __restrict__ last16, const _Float16* __restrict__ w1p,
    const float* __restrict__ b1, const _Float16* __restrict__ w2p,
    const float* __restrict__ b2, float* __restrict__ out) {
  __shared__ __align__(16) _Float16 a16[16][RH];
  __shared__ __align__(16) _Float16 hid16[16][RH];
  const int b0 = blockIdx.x * 16;
  const int tid = threadIdx.x, wave = tid >> 5, lane = tid & 31;
  const int mb = (lane < 16) ? 0 : 8, nn = lane & 15;

  ((uint4*)(&a16[0][0]))[tid] = ((const uint4*)(last16 + (size_t)b0 * RH))[tid];
  __syncthreads();

  // hid = relu(last @ W1^T + b1); wave w -> N-tile w (H = 128 -> 8 tiles)
  v8f acc = {};
#pragma unroll
  for (int kt = 0; kt < 4; ++kt) {
    v16h a = load_a16(&a16[0][0] + kt * 32, RH);
    v16h b = load_b16(w1p + (size_t)(wave * 4 + kt) * 512);
    acc = wmma16(a, b, acc);
  }
  const int n = wave * 16 + nn;
  const float bias = b1[n];
#pragma unroll
  for (int v = 0; v < 8; ++v) {
    float hv = acc[v] + bias;
    hv = hv > 0.0f ? hv : 0.0f;
    hid16[mb + v][n] = (_Float16)hv;
  }
  __syncthreads();

  if (wave < 2) {  // OUT_DIM = 32 -> 2 N-tiles
    v8f a2 = {};
#pragma unroll
    for (int kt = 0; kt < 4; ++kt) {
      v16h a = load_a16(&hid16[0][0] + kt * 32, RH);
      v16h b = load_b16(w2p + (size_t)(wave * 4 + kt) * 512);
      a2 = wmma16(a, b, a2);
    }
    const int n2 = wave * 16 + nn;
    const float bias2 = b2[n2];
#pragma unroll
    for (int v = 0; v < 8; ++v)
      out[(size_t)(b0 + mb + v) * ROUT + n2] = a2[v] + bias2;
  }
}

// -------------------------------------------------------------------------
extern "C" void kernel_launch(void* const* d_in, const int* in_sizes, int n_in,
                              void* d_out, int out_size, void* d_ws,
                              size_t ws_size, hipStream_t stream) {
  (void)in_sizes; (void)n_in; (void)out_size; (void)ws_size;
  const float* x     = (const float*)d_in[0];   // [B,T,64]
  const float* in_W  = (const float*)d_in[1];   // [128,64]
  const float* in_b  = (const float*)d_in[2];   // [128]
  const float* W_ih  = (const float*)d_in[3];   // [4,512,128]
  const float* W_hh  = (const float*)d_in[4];   // [4,512,128]
  const float* b_ih  = (const float*)d_in[5];   // [4,512]
  const float* b_hh  = (const float*)d_in[6];   // [4,512]
  const float* ln_g  = (const float*)d_in[7];   // [4,128]
  const float* ln_b  = (const float*)d_in[8];   // [4,128]
  const float* outW1 = (const float*)d_in[9];   // [128,128]
  const float* outb1 = (const float*)d_in[10];  // [128]
  const float* outW2 = (const float*)d_in[11];  // [32,128]
  const float* outb2 = (const float*)d_in[12];  // [32]
  float* out = (float*)d_out;

  char* base = (char*)d_ws;
  size_t off = 0;
  auto take = [&](size_t bytes) -> void* {
    void* p = base + off;
    off += (bytes + 255) & ~(size_t)255;
    return p;
  };
  _Float16* xp    = (_Float16*)take((size_t)RT * RB * RH * 2);       // 128 MB
  _Float16* wihp  = (_Float16*)take((size_t)RL * 4 * RH * RH * 2);   // 512 KB
  _Float16* whhp  = (_Float16*)take((size_t)RL * 4 * RH * RH * 2);   // 512 KB
  _Float16* inWp  = (_Float16*)take((size_t)RH * RIN * 2);
  _Float16* w1p   = (_Float16*)take((size_t)RH * RH * 2);
  _Float16* w2p   = (_Float16*)take((size_t)ROUT * RH * 2);
  float*    bsum  = (float*)take((size_t)RL * 4 * RH * 4);
  _Float16* last16 = (_Float16*)take((size_t)RB * RH * 2);

  // Pack all weight matrices into WMMA B-fragment blobs (f16).
  hipLaunchKernelGGL(pack_b_kernel, dim3((RH / 16) * (RIN / 32)), dim3(256), 0,
                     stream, in_W, inWp, RH, RIN);
  hipLaunchKernelGGL(pack_b_kernel, dim3((RL * 4 * RH / 16) * (RH / 32)),
                     dim3(256), 0, stream, W_ih, wihp, RL * 4 * RH, RH);
  hipLaunchKernelGGL(pack_b_kernel, dim3((RL * 4 * RH / 16) * (RH / 32)),
                     dim3(256), 0, stream, W_hh, whhp, RL * 4 * RH, RH);
  hipLaunchKernelGGL(pack_b_kernel, dim3((RH / 16) * (RH / 32)), dim3(256), 0,
                     stream, outW1, w1p, RH, RH);
  hipLaunchKernelGGL(pack_b_kernel, dim3((ROUT / 16) * (RH / 32)), dim3(256), 0,
                     stream, outW2, w2p, ROUT, RH);
  hipLaunchKernelGGL(bias_sum_kernel, dim3((RL * 4 * RH + 255) / 256), dim3(256),
                     0, stream, b_ih, b_hh, bsum, RL * 4 * RH);

  // xp = einsum('bti,hi->tbh', x, in_W) + in_b  (f16)
  hipLaunchKernelGGL(inproj_kernel, dim3(RT * RB / 16), dim3(256), 0, stream,
                     x, inWp, in_b, xp);

  // Persistent scan: 32 workgroups, each owns 16 batch rows for all T steps.
  hipLaunchKernelGGL(lstm_scan_kernel, dim3(RB / 16), dim3(256), 0, stream,
                     xp, wihp, whhp, bsum, ln_g, ln_b, last16);

  // Output head.
  hipLaunchKernelGGL(head_kernel, dim3(RB / 16), dim3(256), 0, stream,
                     last16, w1p, outb1, w2p, outb2, out);
}
